// LearnableEdgeMaskplus_78735340470802
// MI455X (gfx1250) — matrix operand order repaired and checked by reading, not compiled, not run on hardware
//
#include <hip/hip_runtime.h>
#include <hip/hip_bf16.h>

// ---------------------------------------------------------------------------
// LearnableEdgeMask pipeline for MI455X (gfx1250, wave32, WMMA).
//   Stage 1: CSR build (counts -> scan -> scatter)           [int atomics only]
//   Stage 2: fused per-16-edge tile: CSR mean-aggregate in regs ->
//            bf16 LDS tiles *pre-swizzled into WMMA fragment order* ->
//            v_wmma_f32_16x16x32_bf16 MLP -> sigmoid probs
//   Stage 3: exact top-k via 2-level radix select on float bit patterns,
//            deterministic lowest-index tie-break (single wave ballot scan)
//   Stage 4: masked incidence outputs
// ---------------------------------------------------------------------------

typedef __bf16 bf16;
typedef __attribute__((ext_vector_type(16))) __bf16 v16bf;
typedef __attribute__((ext_vector_type(4)))  __bf16 v4bf;
typedef __attribute__((ext_vector_type(8)))  float  v8f;

#define DFEAT 128
#define HDIM  32

// ---------------------------------------------------------------- utilities
__global__ void lem_zero_kernel(int* __restrict__ p, int n) {
  int i = blockIdx.x * blockDim.x + threadIdx.x;
  if (i < n) p[i] = 0;
}

__global__ void lem_count_kernel(const int* __restrict__ Eidx,
                                 int* __restrict__ cnt, int nnz) {
  int i = blockIdx.x * blockDim.x + threadIdx.x;
  if (i < nnz) atomicAdd(&cnt[Eidx[i]], 1);
}

// single-block inclusive scan (chunked Hillis-Steele), offsets[0]=0
__global__ __launch_bounds__(1024) void lem_scan_kernel(
    const int* __restrict__ cnt, int* __restrict__ offs, int n) {
  __shared__ int tmp[1024];
  __shared__ int carry;
  if (threadIdx.x == 0) carry = 0;
  __syncthreads();
  for (int base = 0; base < n; base += 1024) {
    int i = base + (int)threadIdx.x;
    int v = (i < n) ? cnt[i] : 0;
    tmp[threadIdx.x] = v;
    __syncthreads();
    for (int off = 1; off < 1024; off <<= 1) {
      int t = (threadIdx.x >= (unsigned)off) ? tmp[threadIdx.x - off] : 0;
      __syncthreads();
      tmp[threadIdx.x] += t;
      __syncthreads();
    }
    int inc = tmp[threadIdx.x] + carry;
    if (i < n) offs[i + 1] = inc;
    __syncthreads();
    if (threadIdx.x == 1023) carry = inc;
    __syncthreads();
  }
  if (threadIdx.x == 0) offs[0] = 0;
}

__global__ void lem_scatter_kernel(const int* __restrict__ Vidx,
                                   const int* __restrict__ Eidx,
                                   const int* __restrict__ offs,
                                   int* __restrict__ cursor,
                                   int* __restrict__ csr_v, int nnz) {
  int i = blockIdx.x * blockDim.x + threadIdx.x;
  if (i < nnz) {
    int e = Eidx[i];
    int p = atomicAdd(&cursor[e], 1);
    csr_v[offs[e] + p] = Vidx[i];
  }
}

// ------------------------------------------------- fused aggregate + WMMA MLP
// one wave32 per block; block handles 16 edges.
// LDS tiles are laid out in WMMA *fragment order*: frag[laneslot][16 bf16],
// so every operand fragment is one contiguous 32B run per lane (2x ds_load_b128)
// with no v_mov_b16 packing.
__global__ __launch_bounds__(32) void lem_edge_mlp_wmma_kernel(
    const float* __restrict__ X, const int* __restrict__ csr_v,
    const int* __restrict__ offs, const float* __restrict__ W1,
    const float* __restrict__ b1, const float* __restrict__ W2,
    const float* __restrict__ b2, float* __restrict__ probs, int nedges) {
  // A fragments: 4 chunks x 32 laneslots x 16 bf16  (16x128 edge-feature tile)
  __shared__ alignas(32) bf16 af[4 * 32 * 16];
  // B fragments: 2 tiles x 4 chunks x 32 laneslots x 16 bf16 (W1 as 128x32 bf16)
  __shared__ alignas(32) bf16 w1f[8 * 32 * 16];
  __shared__ float hbuf[16][HDIM + 1];    // relu hidden, for the final dot

  const int lane = threadIdx.x;
  const int m = lane & 15;
  const int g = lane >> 4;
  const int e0 = blockIdx.x * 16;

  // --- stage W1 -> bf16 LDS directly in B fragment order ------------------
  // B 32x16 (KxN) 16-bit layout: laneslot ls holds col n=ls&15;
// ls<16: elements = K 0..7,16..23 ; ls>=16: elements = K 8..15,24..31.
  for (int t = lane; t < 8 * 32 * 16; t += 32) {
    int el   = t & 15;
    int ls   = (t >> 4) & 31;
    int fc   = t >> 9;            // 0..7 = tile*4 + chunk
    int c    = fc & 3;
    int tile = fc >> 2;
    int kk   = ((el >> 3) << 4) + ((ls >> 4) << 3) + (el & 7);
    int k    = c * 32 + kk;
    int n    = tile * 16 + (ls & 15);
    w1f[t] = (bf16)W1[k * HDIM + n];
  }

  // --- where this lane's 4 contiguous feature cols land in A frag order ---
  // A 16x32 (MxK) 16-bit layout: laneslot ls holds row m=ls&15;
  // ls<16: elements = K 0..7,16..23 ; ls>=16: elements = K 8..15,24..31.
  const int cA  = lane >> 3;           // K-chunk of cols 4*lane..4*lane+3
  const int kkb = (lane & 7) * 4;      // kk within chunk
  int gA, elb;
  if (kkb < 8)       { gA = 0; elb = kkb; }
  else if (kkb < 16) { gA = 1; elb = kkb - 8; }
  else if (kkb < 24) { gA = 0; elb = kkb - 8; }
  else               { gA = 1; elb = kkb - 16; }

  // --- CSR mean aggregation: lane owns 4 contiguous cols (float4 gather) --
  // e is wave-uniform per iteration -> no EXEC divergence around WMMA.
  for (int r = 0; r < 16; ++r) {
    int e = e0 + r;
    float4 acc = make_float4(0.f, 0.f, 0.f, 0.f);
    float inv = 0.f;
    if (e < nedges) {
      int beg = offs[e], end = offs[e + 1];
      for (int j = beg; j < end; ++j) {
        int v = csr_v[j];
        float4 x = ((const float4*)(X + (size_t)v * DFEAT))[lane];
        acc.x += x.x; acc.y += x.y; acc.z += x.z; acc.w += x.w;
      }
      inv = 1.f / fmaxf((float)(end - beg), 1.f);
    }
    v4bf pk;
    pk[0] = (bf16)(acc.x * inv);
    pk[1] = (bf16)(acc.y * inv);
    pk[2] = (bf16)(acc.z * inv);
    pk[3] = (bf16)(acc.w * inv);
    *(v4bf*)&af[((cA * 32) + gA * 16 + r) * 16 + elb] = pk;
  }
  __syncthreads();

  // --- D = A(16x128) * W1(128x32): 4 K-chunks x 2 N-tiles of bf16 WMMA ----
  const v16bf* ap = (const v16bf*)af;
  const v16bf* bp = (const v16bf*)w1f;
  v8f c0 = {};
  v8f c1 = {};
#pragma unroll
  for (int c = 0; c < 4; ++c) {
    v16bf a  = ap[c * 32 + lane];
    v16bf bA = bp[(0 * 4 + c) * 32 + lane];
    v16bf bB = bp[(1 * 4 + c) * 32 + lane];
    c0 = __builtin_amdgcn_wmma_f32_16x16x32_bf16(false, a, false, bA,
                                                 (short)0, c0, false, false);
    c1 = __builtin_amdgcn_wmma_f32_16x16x32_bf16(false, a, false, bB,
                                                 (short)0, c1, false, false);
  }

  // bias + relu; C/D layout: vgpr r holds row (r + 8*g), col = lane&15 (+tile*16)
  float bias0 = b1[m], bias1 = b1[16 + m];
#pragma unroll
  for (int r = 0; r < 8; ++r) {
    int row = r + 8 * g;
    hbuf[row][m]      = fmaxf(c0[r] + bias0, 0.f);
    hbuf[row][16 + m] = fmaxf(c1[r] + bias1, 0.f);
  }
  __syncthreads();

  // logits -> sigmoid (lanes 0..15, one edge each)
  if (lane < 16) {
    int e = e0 + lane;
    if (e < nedges) {
      float s = b2[0];
#pragma unroll
      for (int n = 0; n < HDIM; ++n) s += hbuf[lane][n] * W2[n];
      probs[e] = 1.f / (1.f + __expf(-s));
    }
  }
}

// ----------------------------------------------------------- radix top-k
__global__ void lem_hist_hi_kernel(const float* __restrict__ probs,
                                   unsigned* __restrict__ hist, int n) {
  int i = blockIdx.x * blockDim.x + threadIdx.x;
  if (i < n) {
    unsigned key = __float_as_uint(probs[i]);  // probs > 0 => bits sortable
    atomicAdd(&hist[key >> 16], 1u);
  }
}

__global__ void lem_select_hi_kernel(const unsigned* __restrict__ hist,
                                     unsigned* __restrict__ sel, int topk) {
  if (threadIdx.x == 0 && blockIdx.x == 0) {
    long long acc = 0;
    unsigned bin = 0, above = 0;
    for (int b = 65535; b >= 0; --b) {
      long long h = (long long)hist[b];
      if (acc + h >= (long long)topk) { bin = (unsigned)b; above = (unsigned)acc; break; }
      acc += h;
    }
    sel[0] = bin;
    sel[1] = above;
  }
}

__global__ void lem_hist_lo_kernel(const float* __restrict__ probs,
                                   const unsigned* __restrict__ sel,
                                   unsigned* __restrict__ hist, int n) {
  int i = blockIdx.x * blockDim.x + threadIdx.x;
  if (i < n) {
    unsigned key = __float_as_uint(probs[i]);
    if ((key >> 16) == sel[0]) atomicAdd(&hist[key & 0xFFFFu], 1u);
  }
}

__global__ void lem_select_lo_kernel(const unsigned* __restrict__ hist,
                                     unsigned* __restrict__ sel, int topk) {
  if (threadIdx.x == 0 && blockIdx.x == 0) {
    long long acc = (long long)sel[1];
    for (int b = 65535; b >= 0; --b) {
      long long h = (long long)hist[b];
      if (acc + h >= (long long)topk) {
        sel[2] = (sel[0] << 16) | (unsigned)b;      // exact threshold key
        sel[3] = (unsigned)((long long)topk - acc); // #ties to keep at key T
        break;
      }
      acc += h;
    }
  }
}

__global__ void lem_mark_kernel(const float* __restrict__ probs,
                                const unsigned* __restrict__ sel,
                                int* __restrict__ keep, int n) {
  int i = blockIdx.x * blockDim.x + threadIdx.x;
  if (i < n) {
    unsigned key = __float_as_uint(probs[i]);
    unsigned T = sel[2];
    keep[i] = (key > T) ? 1 : ((key == T) ? 2 : 0);  // 2 = pending tie
  }
}

// single-wave deterministic tie resolution (lowest index first, like lax.top_k)
__global__ __launch_bounds__(32) void lem_resolve_kernel(
    int* __restrict__ keep, const unsigned* __restrict__ sel, int n) {
  const int lane = threadIdx.x;
  const int need = (int)sel[3];
  int granted = 0;
  for (int base = 0; base < n; base += 32) {
    int i = base + lane;
    int pend = (i < n && keep[i] == 2) ? 1 : 0;
    unsigned long long msk = __ballot(pend);
    int prefix = __popcll(msk & ((1ull << lane) - 1ull));
    if (pend) keep[i] = (granted + prefix < need) ? 1 : 0;
    granted += __popcll(msk);
  }
}

// ----------------------------------------------------------------- outputs
__global__ void lem_finalize_kernel(const int* __restrict__ Vidx,
                                    const int* __restrict__ Eidx,
                                    const int* __restrict__ keep,
                                    float* __restrict__ out, int nnz, int E) {
  int i = blockIdx.x * blockDim.x + threadIdx.x;
  if (i >= nnz) return;
  int e = Eidx[i];
  int m = keep[e];
  out[i]                       = m ? (float)Vidx[i] : -1.0f;  // stacked row 0
  out[(size_t)nnz + i]         = m ? (float)e       : -1.0f;  // stacked row 1
  out[2 * (size_t)nnz + E + i] = m ? 1.0f           : 0.0f;   // mask
}

// ---------------------------------------------------------------------------
extern "C" void kernel_launch(void* const* d_in, const int* in_sizes, int n_in,
                              void* d_out, int out_size, void* d_ws, size_t ws_size,
                              hipStream_t stream) {
  const float* X   = (const float*)d_in[0];
  const int* Vidx  = (const int*)d_in[1];
  const int* Eidx  = (const int*)d_in[2];
  const float* W1  = (const float*)d_in[3];
  const float* b1  = (const float*)d_in[4];
  const float* W2  = (const float*)d_in[5];
  const float* b2  = (const float*)d_in[6];

  const int nnz  = in_sizes[1];               // 1,600,000
  const int E    = out_size - 3 * nnz;        // 100,000
  const int topk = E / 2;

  // workspace layout (~8.5 MB)
  int* cnt        = (int*)d_ws;               // [E]
  int* cursor     = cnt + E;                  // [E]
  int* offs       = cursor + E;               // [E+1]
  int* csr_v      = offs + (E + 1);           // [nnz]
  unsigned* hist1 = (unsigned*)(csr_v + nnz); // [65536]
  unsigned* hist2 = hist1 + 65536;            // [65536]
  unsigned* sel   = hist2 + 65536;            // [16]
  int* keep       = (int*)(sel + 16);         // [E]

  float* out   = (float*)d_out;
  float* probs = out + 2 * (size_t)nnz;       // probs live inside d_out

  const int B = 256;

  // zero: cnt+cursor (contiguous), hist1+hist2+sel (contiguous)
  lem_zero_kernel<<<(2 * E + B - 1) / B, B, 0, stream>>>(cnt, 2 * E);
  lem_zero_kernel<<<(2 * 65536 + 16 + B - 1) / B, B, 0, stream>>>(
      (int*)hist1, 2 * 65536 + 16);

  // CSR build
  lem_count_kernel<<<(nnz + B - 1) / B, B, 0, stream>>>(Eidx, cnt, nnz);
  lem_scan_kernel<<<1, 1024, 0, stream>>>(cnt, offs, E);
  lem_scatter_kernel<<<(nnz + B - 1) / B, B, 0, stream>>>(Vidx, Eidx, offs,
                                                          cursor, csr_v, nnz);

  // fused mean-aggregate + WMMA MLP + sigmoid
  lem_edge_mlp_wmma_kernel<<<(E + 15) / 16, 32, 0, stream>>>(
      X, csr_v, offs, W1, b1, W2, b2, probs, E);

  // exact top-k (2-level radix select on float bit patterns)
  lem_hist_hi_kernel<<<(E + B - 1) / B, B, 0, stream>>>(probs, hist1, E);
  lem_select_hi_kernel<<<1, 32, 0, stream>>>(hist1, sel, topk);
  lem_hist_lo_kernel<<<(E + B - 1) / B, B, 0, stream>>>(probs, sel, hist2, E);
  lem_select_lo_kernel<<<1, 32, 0, stream>>>(hist2, sel, topk);
  lem_mark_kernel<<<(E + B - 1) / B, B, 0, stream>>>(probs, sel, keep, E);
  lem_resolve_kernel<<<1, 32, 0, stream>>>(keep, sel, E);

  // masked incidence outputs + mask
  lem_finalize_kernel<<<(nnz + B - 1) / B, B, 0, stream>>>(Vidx, Eidx, keep,
                                                           out, nnz, E);
}
